// HMM_16174846837041
// MI455X (gfx1250) — matrix operand order repaired and checked by reading, not compiled, not run on hardware
//
#include <hip/hip_runtime.h>
#include <stdint.h>

// ---------------------------------------------------------------------------
// HMM forward recursion on MI455X (gfx1250, wave32, WMMA).
//   alpha_{t} = (alpha_{t-1} @ A) * B[:, obs[t]]   for t = 1..1023
//   out = sum(alpha_1023)
// A (4096x4096) is converted once to bf16 (32 MB, L2-resident) in
// v_wmma_f32_16x16x32_bf16 B-fragment layout, K = 16*half+s per lane. A persistent cooperative kernel
// runs all 1023 steps with one device-wide barrier per step.
// ---------------------------------------------------------------------------

#define N_STATES 4096
#define N_OBS    128
#define SEQ_LEN  1024
#define N_CB     256        // column blocks of 16
#define N_KC     128        // K chunks of 32
#define TILE_BYTES 1024     // one 32x16 bf16 tile in fragment order
#define GRID_MAIN 128       // persistent blocks (2 column blocks each)

typedef __attribute__((ext_vector_type(16))) __bf16 v16bf;
typedef __attribute__((ext_vector_type(8)))  float  v8f;

union Frag {
    v16bf bf;
    uint4 q[2];
};

__device__ __forceinline__ uint16_t f2bf(float f) {
    union { float f; uint32_t u; } c; c.f = f;
    uint32_t u = c.u;
    uint32_t r = u + 0x7FFFu + ((u >> 16) & 1u);   // round-to-nearest-even
    if ((u & 0x7F800000u) == 0x7F800000u) r = u;   // keep NaN/Inf
    return (uint16_t)(r >> 16);
}

// ---------------------------------------------------------------------------
// Pack A (fp32 row-major, alpha@A => A[i][j] at i*4096+j) into bf16 tiles in
// WMMA B-fragment layout with K = 16*half+s. Thread <-> (cb, kc, lane): produces the lane's
// 32 contiguous bytes (8 dwords = K-pairs 2v,2v+1 of one column).
// Reads coalesced across columns; writes fully coalesced b128.
// ---------------------------------------------------------------------------
__global__ __launch_bounds__(256) void hmm_pack(const float* __restrict__ A,
                                                uint8_t* __restrict__ packed) {
    unsigned tid  = blockIdx.x * 256u + threadIdx.x;  // 1,048,576 threads
    unsigned L    = tid & 31u;
    unsigned kc   = (tid >> 5) & (N_KC - 1u);
    unsigned cb   = tid >> 12;
    unsigned c    = L & 15u;
    unsigned half = L >> 4;                           // lanes 16-31 -> K 16..31
    unsigned col  = cb * 16u + c;
    unsigned krow = kc * 32u + half * 16u;

    uint32_t u[8];
#pragma unroll
    for (int v = 0; v < 8; ++v) {
        unsigned k = krow + 2u * (unsigned)v;
        uint16_t lo = f2bf(A[(size_t)k       * N_STATES + col]);
        uint16_t hi = f2bf(A[(size_t)(k + 1) * N_STATES + col]);
        u[v] = (uint32_t)lo | ((uint32_t)hi << 16);
    }
    uint4* dst = (uint4*)(packed + (size_t)tid * 32u);
    dst[0] = make_uint4(u[0], u[1], u[2], u[3]);
    dst[1] = make_uint4(u[4], u[5], u[6], u[7]);
}

// alpha0[j] = pi[j] * B[j, obs[0]]  (stored bf16, buffer 0)
__global__ __launch_bounds__(256) void hmm_init(const float* __restrict__ pi,
                                                const float* __restrict__ B,
                                                const int* __restrict__ obs,
                                                uint16_t* __restrict__ alphaBuf) {
    int j = blockIdx.x * 256 + threadIdx.x;
    if (j < N_STATES) {
        int o = obs[0];
        alphaBuf[j] = f2bf(pi[j] * B[(size_t)j * N_OBS + o]);
    }
}

// ---------------------------------------------------------------------------
// Persistent main kernel. 128 blocks x 256 threads (8 waves). Each block owns
// 2 column blocks; 4 waves per column block split K=4096 into quarters.
// Per step per wave: 32 x (ds b128 x2 for replicated alpha A-frag,
//                          global b128 x2 for matrix B-frag, 1 WMMA).
// ---------------------------------------------------------------------------
__global__ __launch_bounds__(256) void hmm_main(const float* __restrict__ B,
                                                const int* __restrict__ obs,
                                                const uint8_t* __restrict__ packedA,
                                                uint16_t* __restrict__ alphaBuf,
                                                float* __restrict__ cbsums,
                                                unsigned* __restrict__ cnt,
                                                float* __restrict__ out) {
    __shared__ uint16_t s_alpha[N_STATES];   // 8 KB current alpha (bf16)
    __shared__ float    s_part[8 * 32];      // per-wave partial row
    __shared__ float    s_fin[32];           // final-step values per block

    const unsigned tidx    = threadIdx.x;
    const unsigned lane    = tidx & 31u;
    const unsigned wave    = tidx >> 5;
    const unsigned half    = lane >> 4;
    const unsigned cbLocal = wave >> 2;          // 0 or 1
    const unsigned q       = wave & 3u;          // K quarter
    const unsigned myCb    = blockIdx.x * 2u + cbLocal;
    const unsigned nBlocks = gridDim.x;
    const uint8_t* myTiles = packedA + ((size_t)myCb * N_KC + (size_t)q * 32u) * TILE_BYTES;

    for (unsigned t = 1; t < SEQ_LEN; ++t) {
        const uint16_t* cur = alphaBuf + ((t - 1u) & 1u) * N_STATES;
        uint16_t*       nxt = alphaBuf + (t & 1u) * N_STATES;

        // Stage alpha into LDS (512 uint4, 2 per thread)
        {
            const uint4* src = (const uint4*)cur;
            uint4*       dst = (uint4*)s_alpha;
            dst[tidx]        = src[tidx];
            dst[tidx + 256]  = src[tidx + 256];
        }
        __syncthreads();

        v8f acc = {};
        const uint32_t* aw = (const uint32_t*)s_alpha;
        const uint8_t*  tp = myTiles;
#pragma unroll 4
        for (unsigned kcl = 0; kcl < 32u; ++kcl) {
            unsigned kc = q * 32u + kcl;
            // A-fragment: alpha replicated across all 16 M rows.
            // Lane-half K contiguous (matches packed B):
            //   lanes 0-15: dwords {0..7}; lanes 16-31: {8..15}
            const uint4* aq = (const uint4*)(aw + kc * 16u);
            Frag a, b;
            a.q[0] = aq[2u * half];
            a.q[1] = aq[2u * half + 1u];
            // B-fragment: pre-packed tile, lane-contiguous 32 bytes.
            const uint4* bq = (const uint4*)(tp + (size_t)kcl * TILE_BYTES + (size_t)lane * 32u);
            b.q[0] = bq[0];
            b.q[1] = bq[1];
            acc = __builtin_amdgcn_wmma_f32_16x16x32_bf16(
                false, a.bf, false, b.bf, (short)0, acc, false, false);
        }

        // Rows are replicated => acc[0] on lanes 0-15 is the result for col N=lane.
        s_part[wave * 32u + lane] = acc[0];
        __syncthreads();

        if (q == 0u && lane < 16u) {
            unsigned base = (cbLocal * 4u) * 32u + lane;
            float s = s_part[base];
            s += s_part[base + 32u];
            s += s_part[base + 64u];
            s += s_part[base + 96u];
            unsigned col = myCb * 16u + lane;
            int o = obs[t];
            float val = s * B[(size_t)col * N_OBS + o];   // emission
            nxt[col] = f2bf(val);
            if (t == SEQ_LEN - 1u) s_fin[cbLocal * 16u + lane] = val;
        }
        if (t == SEQ_LEN - 1u) {
            __syncthreads();
            if (tidx < 2u) {                      // deterministic ordered sum
                float s = 0.f;
                for (int i = 0; i < 16; ++i) s += s_fin[tidx * 16u + i];
                cbsums[blockIdx.x * 2u + tidx] = s;
            }
        }

        // -------- device-wide barrier (monotone counter, one add/block) ----
        __syncthreads();
        if (tidx == 0u) {
            __threadfence();
            atomicAdd(cnt, 1u);
            unsigned target = t * nBlocks;
            while (__hip_atomic_load(cnt, __ATOMIC_RELAXED,
                                     __HIP_MEMORY_SCOPE_AGENT) < target) {
                __builtin_amdgcn_s_sleep(1);
            }
            __threadfence();
        }
        __syncthreads();
    }

    if (blockIdx.x == 0u && tidx == 0u) {        // deterministic final sum
        float s = 0.f;
        for (int i = 0; i < N_CB; ++i) s += cbsums[i];
        out[0] = s;
    }
}

extern "C" void kernel_launch(void* const* d_in, const int* in_sizes, int n_in,
                              void* d_out, int out_size, void* d_ws, size_t ws_size,
                              hipStream_t stream) {
    (void)in_sizes; (void)n_in; (void)out_size;

    const float* A   = (const float*)d_in[0];
    const float* B   = (const float*)d_in[1];
    const float* pi  = (const float*)d_in[2];
    const int*   obs = (const int*)d_in[3];

    uint8_t* ws = (uint8_t*)d_ws;
    const size_t packedBytes = (size_t)N_CB * N_KC * TILE_BYTES;   // 32 MB
    uint8_t*  packedA  = ws;
    uint16_t* alphaBuf = (uint16_t*)(ws + packedBytes);            // 2 x 4096 bf16
    float*    cbsums   = (float*)(ws + packedBytes + 2u * N_STATES * 2u);
    unsigned* cnt      = (unsigned*)(ws + packedBytes + 2u * N_STATES * 2u + N_CB * 4u);
    const size_t needed = packedBytes + 2u * N_STATES * 2u + N_CB * 4u + 64u;
    if (ws_size < needed) return;   // scratch too small to run safely

    hipMemsetAsync(cnt, 0, sizeof(unsigned), stream);              // reset barrier
    hmm_pack<<<4096, 256, 0, stream>>>(A, packedA);
    hmm_init<<<(N_STATES + 255) / 256, 256, 0, stream>>>(pi, B, obs, alphaBuf);
    hmm_main<<<GRID_MAIN, 256, 0, stream>>>(B, obs, packedA, alphaBuf,
                                            cbsums, cnt, (float*)d_out);
}
